// Net_88459146429015
// MI455X (gfx1250) — matrix operand order repaired and checked by reading, not compiled, not run on hardware
//
#include <hip/hip_runtime.h>
#include <hip/hip_bf16.h>

// ---------------- problem constants ----------------
#define BB   128      // batch
#define DD   128      // input dim
#define TT   2000     // timesteps
#define HH   512      // hidden
#define OO   7        // output dim
#define LD0  (DD + HH)        // 640  : layer-0 concat K
#define LD1  (HH + HH)        // 1024 : layer-1 concat K
#define G4H  (4 * HH)         // 2048 : gate rows

// persistent grid: 8 M-tiles x 32 J-tiles workgroups, 4 waves (gate chunks) each
#define NBLK 256
#define NTHR 128

// ---------------- workspace layout (bytes) ----------------
static constexpr size_t OFF_W0  = 0;                                   // bf16 [2048,640]
static constexpr size_t OFF_W1  = OFF_W0  + (size_t)G4H * LD0 * 2;     // bf16 [2048,1024]
static constexpr size_t OFF_B0  = OFF_W1  + (size_t)G4H * LD1 * 2;     // f32  [2048]
static constexpr size_t OFF_B1  = OFF_B0  + (size_t)G4H * 4;           // f32  [2048]
static constexpr size_t OFF_IN0 = OFF_B1  + (size_t)G4H * 4;           // bf16 [128,640]
static constexpr size_t OFF_IN1 = OFF_IN0 + (size_t)BB * LD0 * 2;      // bf16 [128,1024]
static constexpr size_t OFF_H1F = OFF_IN1 + (size_t)BB * LD1 * 2;      // f32  [128,512]
static constexpr size_t OFF_BAR = OFF_H1F + (size_t)BB * HH * 4;       // 2 x u32

// ---------------- types ----------------
typedef __attribute__((ext_vector_type(16))) __bf16    v16bf;
typedef __attribute__((ext_vector_type(8)))  float     v8f;
typedef __attribute__((ext_vector_type(4)))  unsigned  v4u;

union FragBF { v4u u[2]; v16bf v; };

// fast activations: v_exp_f32 + v_rcp_f32 (keep the serial tail off the IEEE-div path)
__device__ __forceinline__ float sig_(float x) {
  return __builtin_amdgcn_rcpf(1.f + __expf(-x));
}
__device__ __forceinline__ float tanh_(float x) {
  // tanh(x) = 1 - 2/(exp(2x)+1)
  float e = __expf(2.f * x);
  return 1.f - 2.f * __builtin_amdgcn_rcpf(e + 1.f);
}
__device__ __forceinline__ unsigned pack2bf(float a, float b) {
  unsigned short ua = __builtin_bit_cast(unsigned short, (__bf16)a);
  unsigned short ub = __builtin_bit_cast(unsigned short, (__bf16)b);
  return ((unsigned)ub << 16) | (unsigned)ua;
}

// ---------------- grid-wide barrier (all NBLK blocks resident) ----------------
__device__ __forceinline__ void grid_barrier(volatile unsigned* cnt,
                                             volatile unsigned* gen) {
  __syncthreads();
  if (threadIdx.x == 0) {
    __threadfence();
    unsigned g = *gen;                       // read generation BEFORE arriving
    if (atomicAdd((unsigned*)cnt, 1u) == NBLK - 1u) {
      *cnt = 0u;
      __threadfence();
      atomicExch((unsigned*)gen, g + 1u);    // release everyone
    } else {
      while (*gen == g) { __builtin_amdgcn_s_sleep(1); }
    }
    __threadfence();
  }
  __syncthreads();
}

// ---------------- prep: pack weights to bf16, fuse biases, init state ----------------
__global__ void k_prep(const float* __restrict__ x,
                       const float* __restrict__ Wih0, const float* __restrict__ Whh0,
                       const float* __restrict__ bih0, const float* __restrict__ bhh0,
                       const float* __restrict__ Wih1, const float* __restrict__ Whh1,
                       const float* __restrict__ bih1, const float* __restrict__ bhh1,
                       __bf16* W0, __bf16* W1, float* bias0, float* bias1,
                       __bf16* in0, __bf16* in1, unsigned* bar) {
  const int R0 = G4H * LD0;
  const int R1 = R0 + G4H * LD1;
  const int R2 = R1 + G4H;
  const int R3 = R2 + G4H;
  const int R4 = R3 + BB * LD0;
  const int R5 = R4 + BB * LD1;
  const int NT = R5 + 2;
  for (int i = blockIdx.x * blockDim.x + threadIdx.x; i < NT;
       i += gridDim.x * blockDim.x) {
    if (i < R0) {                              // W0cat = [W_ih0 | W_hh0]  row-major [2048,640]
      int n = i / LD0, k = i % LD0;
      float v = (k < DD) ? Wih0[n * DD + k] : Whh0[n * HH + (k - DD)];
      W0[i] = (__bf16)v;
    } else if (i < R1) {                       // W1cat = [W_ih1 | W_hh1]  row-major [2048,1024]
      int ii = i - R0; int n = ii / LD1, k = ii % LD1;
      float v = (k < HH) ? Wih1[n * HH + k] : Whh1[n * HH + (k - HH)];
      W1[ii] = (__bf16)v;
    } else if (i < R2) { int ii = i - R1; bias0[ii] = bih0[ii] + bhh0[ii]; }
    else if (i < R3)   { int ii = i - R2; bias1[ii] = bih1[ii] + bhh1[ii]; }
    else if (i < R4) {                         // in0 = [x_0 | h0=0]
      int ii = i - R3; int b = ii / LD0, col = ii % LD0;
      in0[ii] = (col < DD) ? (__bf16)x[((size_t)b * DD + col) * TT] : (__bf16)0.f;
    } else if (i < R5) { in1[i - R4] = (__bf16)0.f; }   // in1 = [h0 | h1] = 0
    else { bar[i - R5] = 0u; }
  }
}

// ---------------- one 16x16 WMMA tile: acc = A[16xK] * B[Kx16] -----------------
// Dual accumulators: consecutive k-slices go to independent acc chains so the
// back-to-back v_wmma ops pipeline instead of serializing on the C operand.
// A row-major bf16 (lane m = l%16, K halves split by l/16, 2x16B contiguous loads)
// B taken from row-major W[n,k] (lane n = l%16, 16 consecutive k, 2x16B loads)
__device__ __forceinline__ v8f gemm_tile(const __bf16* __restrict__ Arow,
                                         const __bf16* __restrict__ Brow,
                                         int K) {
  v8f acc0 = {}, acc1 = {};
#pragma clang loop unroll_count(2)
  for (int k = 0; k < K; k += 64) {
    FragBF a0, b0, a1, b1;
    a0.u[0] = *(const v4u*)(Arow + k);
    a0.u[1] = *(const v4u*)(Arow + k + 8);
    b0.u[0] = *(const v4u*)(Brow + k);
    b0.u[1] = *(const v4u*)(Brow + k + 8);
    a1.u[0] = *(const v4u*)(Arow + k + 32);
    a1.u[1] = *(const v4u*)(Arow + k + 40);
    b1.u[0] = *(const v4u*)(Brow + k + 32);
    b1.u[1] = *(const v4u*)(Brow + k + 40);
    acc0 = __builtin_amdgcn_wmma_f32_16x16x32_bf16(
        false, a0.v, false, b0.v, (short)0, acc0, false, false);
    acc1 = __builtin_amdgcn_wmma_f32_16x16x32_bf16(
        false, a1.v, false, b1.v, (short)0, acc1, false, false);
  }
  return acc0 + acc1;
}

// ---------------- persistent 2-layer LSTM over all T ----------------
__global__ __launch_bounds__(NTHR, 1) void k_lstm(
    const float* __restrict__ x, const float* __restrict__ Wout,
    const float* __restrict__ bout,
    const __bf16* __restrict__ W0, const __bf16* __restrict__ W1,
    const float* __restrict__ bias0, const float* __restrict__ bias1,
    __bf16* in0, __bf16* in1, float* h1f,
    unsigned* bar, float* out) {
  __shared__ float lds_g[4 * 256];           // 4 gate chunks x 16x16 f32 tiles

  const int tid  = threadIdx.x;
  const int lane = tid & 31;
  const int q    = tid >> 5;                 // gate chunk 0..3 (i,f,g,o)
  const int mt   = blockIdx.x >> 5;          // batch tile 0..7
  const int jt   = blockIdx.x & 31;          // hidden-col tile 0..31
  const int hk   = lane >> 4;                // K-half selector
  const int lm   = lane & 15;

  // elementwise mapping: thread handles tile elements e0, e0+1 (same row)
  const int e0   = tid * 2;
  const int er   = e0 >> 4;                  // tile row 0..15
  const int ec   = e0 & 15;                  // tile col (even)
  const int grow = mt * 16 + er;             // global batch row
  const int j    = jt * 16 + ec;             // hidden column (even)

  // fragment base pointers (per-lane rows)
  const __bf16* A0 = in0 + (size_t)(mt * 16 + lm) * LD0 + hk * 16;
  const __bf16* B0 = W0  + (size_t)(q * HH + jt * 16 + lm) * LD0 + hk * 16;
  const __bf16* A1 = in1 + (size_t)(mt * 16 + lm) * LD1 + hk * 16;
  const __bf16* B1 = W1  + (size_t)(q * HH + jt * 16 + lm) * LD1 + hk * 16;

  volatile unsigned* cnt = bar;
  volatile unsigned* gen = bar + 1;

  // persistent cell state (this workgroup's 16x16 tile, 2 elems/thread/layer)
  float c0a = 0.f, c0b = 0.f, c1a = 0.f, c1b = 0.f;

  const float bi0 = bias0[0 * HH + j],     bi0b = bias0[0 * HH + j + 1];
  const float bf0 = bias0[1 * HH + j],     bf0b = bias0[1 * HH + j + 1];
  const float bg0 = bias0[2 * HH + j],     bg0b = bias0[2 * HH + j + 1];
  const float bo0 = bias0[3 * HH + j],     bo0b = bias0[3 * HH + j + 1];
  const float bi1 = bias1[0 * HH + j],     bi1b = bias1[0 * HH + j + 1];
  const float bf1 = bias1[1 * HH + j],     bf1b = bias1[1 * HH + j + 1];
  const float bg1 = bias1[2 * HH + j],     bg1b = bias1[2 * HH + j + 1];
  const float bo1 = bias1[3 * HH + j],     bo1b = bias1[3 * HH + j + 1];

  for (int t = 0; t < TT; ++t) {
    // ============ phase 1: layer 0  gates0 = [x_t | h0] @ W0^T ============
    {
      v8f acc = gemm_tile(A0, B0, LD0);
      // deposit 16x16 tile: C/D layout lane m = v + 8*(l/16), n = l%16
#pragma unroll
      for (int v = 0; v < 8; ++v)
        lds_g[q * 256 + (v + 8 * hk) * 16 + lm] = acc[v];
    }
    __syncthreads();
    {
      float ia = lds_g[0 * 256 + e0] + bi0, ib = lds_g[0 * 256 + e0 + 1] + bi0b;
      float fa = lds_g[1 * 256 + e0] + bf0, fb = lds_g[1 * 256 + e0 + 1] + bf0b;
      float ga = lds_g[2 * 256 + e0] + bg0, gb = lds_g[2 * 256 + e0 + 1] + bg0b;
      float oa = lds_g[3 * 256 + e0] + bo0, ob = lds_g[3 * 256 + e0 + 1] + bo0b;
      c0a = sig_(fa) * c0a + sig_(ia) * tanh_(ga);
      c0b = sig_(fb) * c0b + sig_(ib) * tanh_(gb);
      float ha = sig_(oa) * tanh_(c0a);
      float hb = sig_(ob) * tanh_(c0b);
      unsigned pk = pack2bf(ha, hb);
      *(unsigned*)(in0 + (size_t)grow * LD0 + DD + j) = pk;  // h0 -> layer0 input (t+1)
      *(unsigned*)(in1 + (size_t)grow * LD1 + j)      = pk;  // h0 -> layer1 input (t)
    }
    grid_barrier(cnt, gen);

    // ============ phase 2: layer 1  gates1 = [h0 | h1] @ W1^T ============
    {
      v8f acc = gemm_tile(A1, B1, LD1);
#pragma unroll
      for (int v = 0; v < 8; ++v)
        lds_g[q * 256 + (v + 8 * hk) * 16 + lm] = acc[v];
    }
    __syncthreads();
    {
      float ia = lds_g[0 * 256 + e0] + bi1, ib = lds_g[0 * 256 + e0 + 1] + bi1b;
      float fa = lds_g[1 * 256 + e0] + bf1, fb = lds_g[1 * 256 + e0 + 1] + bf1b;
      float ga = lds_g[2 * 256 + e0] + bg1, gb = lds_g[2 * 256 + e0 + 1] + bg1b;
      float oa = lds_g[3 * 256 + e0] + bo1, ob = lds_g[3 * 256 + e0 + 1] + bo1b;
      c1a = sig_(fa) * c1a + sig_(ia) * tanh_(ga);
      c1b = sig_(fb) * c1b + sig_(ib) * tanh_(gb);
      float ha = sig_(oa) * tanh_(c1a);
      float hb = sig_(ob) * tanh_(c1b);
      *(unsigned*)(in1 + (size_t)grow * LD1 + HH + j) = pack2bf(ha, hb);
      h1f[(size_t)grow * HH + j]     = ha;               // fp32 copy for output head
      h1f[(size_t)grow * HH + j + 1] = hb;
    }
    // overlap: stream x_{t+1} into in0's x-columns (bf16) while layer-1 settles
    if (t + 1 < TT) {
      int g = blockIdx.x * NTHR + tid;                   // 32768 threads, 16384 elems
      if (g < BB * DD) {
        int b = g >> 7, d = g & 127;
        in0[(size_t)b * LD0 + d] = (__bf16)x[((size_t)b * DD + d) * TT + (t + 1)];
      }
    }
    grid_barrier(cnt, gen);

    // ============ phase 3: output head (jt==0 blocks), overlaps next phase 1 ======
    if (jt == 0 && tid < 16 * OO) {
      int r = tid / OO, o = tid % OO;
      const float4* hrow = (const float4*)(h1f + (size_t)(mt * 16 + r) * HH);
      const float4* wrow = (const float4*)(Wout + (size_t)o * HH);
      float s0 = bout[o], s1 = 0.f, s2 = 0.f, s3 = 0.f;
#pragma clang loop unroll_count(4)
      for (int k2 = 0; k2 < HH / 4; ++k2) {
        float4 hv = hrow[k2];
        float4 wv = wrow[k2];
        s0 += hv.x * wv.x; s1 += hv.y * wv.y;
        s2 += hv.z * wv.z; s3 += hv.w * wv.w;
      }
      out[((size_t)t * BB + (mt * 16 + r)) * OO + o] = (s0 + s1) + (s2 + s3);
    }
  }
}

// ---------------- host-side launch ----------------
extern "C" void kernel_launch(void* const* d_in, const int* in_sizes, int n_in,
                              void* d_out, int out_size, void* d_ws, size_t ws_size,
                              hipStream_t stream) {
  const float* x    = (const float*)d_in[0];
  const float* Wih0 = (const float*)d_in[1];
  const float* Whh0 = (const float*)d_in[2];
  const float* bih0 = (const float*)d_in[3];
  const float* bhh0 = (const float*)d_in[4];
  const float* Wih1 = (const float*)d_in[5];
  const float* Whh1 = (const float*)d_in[6];
  const float* bih1 = (const float*)d_in[7];
  const float* bhh1 = (const float*)d_in[8];
  const float* Wout = (const float*)d_in[9];
  const float* bout = (const float*)d_in[10];

  char* ws = (char*)d_ws;
  __bf16*  W0    = (__bf16*)(ws + OFF_W0);
  __bf16*  W1    = (__bf16*)(ws + OFF_W1);
  float*   bias0 = (float*)(ws + OFF_B0);
  float*   bias1 = (float*)(ws + OFF_B1);
  __bf16*  in0   = (__bf16*)(ws + OFF_IN0);
  __bf16*  in1   = (__bf16*)(ws + OFF_IN1);
  float*   h1f   = (float*)(ws + OFF_H1F);
  unsigned* bar  = (unsigned*)(ws + OFF_BAR);

  k_prep<<<2048, 256, 0, stream>>>(x, Wih0, Whh0, bih0, bhh0,
                                   Wih1, Whh1, bih1, bhh1,
                                   W0, W1, bias0, bias1, in0, in1, bar);

  k_lstm<<<NBLK, NTHR, 0, stream>>>(x, Wout, bout, W0, W1, bias0, bias1,
                                    in0, in1, h1f, bar, (float*)d_out);
}
